// MultiCrossAttention_60636348285630
// MI455X (gfx1250) — compile-verified, matrix operand order checked by reading, not run on hardware
//
#include <hip/hip_runtime.h>
#include <hip/hip_bf16.h>

// ---------------------------------------------------------------------------
// MultiCrossAttention for MI455X (gfx1250): bf16 WMMA pipeline, 2x2 register
// blocking (32x32 output per wave) for 2x arithmetic intensity per vmem load.
// ---------------------------------------------------------------------------

typedef __hip_bfloat16 bf16;
typedef __attribute__((ext_vector_type(16))) __bf16 bfx16;
typedef __attribute__((ext_vector_type(8)))  float  v8f;
typedef __attribute__((ext_vector_type(4)))  unsigned int u32x4;
typedef __attribute__((ext_vector_type(4)))  float  f32x4;

union FragU { bfx16 v; u32x4 q[2]; };

// A/B fragment load per CDNA5 16-bit WMMA layout:
// lane 0-15  hold K = [k0..k0+7]   and [k0+16..k0+23]
// lane 16-31 hold K = [k0+8..k0+15] and [k0+24..k0+31]
__device__ __forceinline__ bfx16 load_frag(const bf16* rowbase, int k0, int hi) {
  FragU f;
  f.q[0] = *(const u32x4*)(rowbase + k0 + hi * 8);
  f.q[1] = *(const u32x4*)(rowbase + k0 + 16 + hi * 8);
  return f.v;
}

__device__ __forceinline__ v8f wmma_bf16(bfx16 a, bfx16 b, v8f c) {
  return __builtin_amdgcn_wmma_f32_16x16x32_bf16(false, a, false, b,
                                                 (short)0, c, false, false);
}

// 32x32 output block per wave: acc[u][v] is tile (m0+16u, n0+16v).
// A[M,K] @ W[N,K]^T, both bf16 row-major. 4 WMMA per 8 fragment loads.
__device__ __forceinline__ void gemm_2x2(const bf16* A, int lda, int m0,
                                         const bf16* W, int ldw, int n0, int K,
                                         v8f acc[2][2]) {
  const int lane = threadIdx.x & 31;
  const int hi   = lane >> 4;
  const int l15  = lane & 15;
  const bf16* ar0 = A + (size_t)(m0 + l15) * lda;
  const bf16* ar1 = A + (size_t)(m0 + 16 + l15) * lda;
  const bf16* wr0 = W + (size_t)(n0 + l15) * ldw;
  const bf16* wr1 = W + (size_t)(n0 + 16 + l15) * ldw;
  for (int k0 = 0; k0 < K; k0 += 32) {
    bfx16 fa0 = load_frag(ar0, k0, hi);
    bfx16 fa1 = load_frag(ar1, k0, hi);
    bfx16 fb0 = load_frag(wr0, k0, hi);
    bfx16 fb1 = load_frag(wr1, k0, hi);
    acc[0][0] = wmma_bf16(fa0, fb0, acc[0][0]);
    acc[0][1] = wmma_bf16(fa0, fb1, acc[0][1]);
    acc[1][0] = wmma_bf16(fa1, fb0, acc[1][0]);
    acc[1][1] = wmma_bf16(fa1, fb1, acc[1][1]);
  }
}

__device__ __forceinline__ float gelu_exact(float v) {
  return 0.5f * v * (1.0f + erff(v * 0.70710678118654752f));
}

// ---------------------------------------------------------------------------
__global__ void convert_f32_bf16(const float* __restrict__ src,
                                 bf16* __restrict__ dst, int n) {
  int i = blockIdx.x * blockDim.x + threadIdx.x;
  if (i < n) dst[i] = __float2bfloat16(src[i]);
}

// AvgPool2d(r,r) on [B, H*W, C] -> bf16 [B*(H/r)*(W/r), C]; float4 reads.
__global__ void pool_kernel(const float* __restrict__ y, bf16* __restrict__ out,
                            int H, int W, int C, int r) {
  int orow = blockIdx.x;             // b*256 + p  (always 256 out pixels)
  int b = orow >> 8;
  int p = orow & 255;
  int OW = W / r;
  int oh = p / OW, ow = p % OW;
  float inv = 1.0f / (float)(r * r);
  for (int c = threadIdx.x * 4; c < C; c += blockDim.x * 4) {
    f32x4 s = {};
    for (int i = 0; i < r; ++i) {
      const float* rowp = y + ((size_t)b * H * W + (size_t)(oh * r + i) * W + (size_t)ow * r) * C + c;
      for (int j = 0; j < r; ++j)
        s += *(const f32x4*)(rowp + (size_t)j * C);
    }
    bf16* o = out + (size_t)orow * C + c;
    o[0] = __float2bfloat16(s.x * inv);
    o[1] = __float2bfloat16(s.y * inv);
    o[2] = __float2bfloat16(s.z * inv);
    o[3] = __float2bfloat16(s.w * inv);
  }
}

// Branch: pooled[M,C] @ sr_w[C,C]^T + b  -> LayerNorm -> GELU -> xc[:, colOff:colOff+C]
// One block = 32 rows x all N columns (so LN sees complete rows).
__global__ void gemm_ln_gelu(const bf16* __restrict__ A, const bf16* __restrict__ Wt,
                             const float* __restrict__ bias,
                             const float* __restrict__ lng, const float* __restrict__ lnb,
                             bf16* __restrict__ xc, int colOff, int N) {
  extern __shared__ float sm[];
  float* tileC = sm;            // [32][N]
  float* stats = sm + 32 * N;   // mean[32], rstd[32]
  const int m0   = blockIdx.x * 32;
  const int wave = threadIdx.x >> 5;
  const int lane = threadIdx.x & 31;
  const int hi   = lane >> 4;
  const int l15  = lane & 15;
  const int nmac = N >> 5;      // 32-col macros (2, 4, 10)
  for (int t = wave; t < nmac; t += 8) {
    v8f acc[2][2] = {};
    gemm_2x2(A, N, m0, Wt, N, t * 32, N, acc);
#pragma unroll
    for (int u = 0; u < 2; ++u)
#pragma unroll
      for (int vv = 0; vv < 2; ++vv)
#pragma unroll
        for (int i = 0; i < 8; ++i) {
          int rr = u * 16 + i + 8 * hi;
          int cc = t * 32 + vv * 16 + l15;
          tileC[rr * N + cc] = acc[u][vv][i] + bias[cc];
        }
  }
  __syncthreads();
  if (threadIdx.x < 32) {
    const float* row = tileC + threadIdx.x * N;
    float s = 0.f, ss = 0.f;
    for (int j = 0; j < N; ++j) { float v = row[j]; s += v; ss += v * v; }
    float mean = s / (float)N;
    float var  = ss / (float)N - mean * mean;
    stats[threadIdx.x]      = mean;
    stats[32 + threadIdx.x] = rsqrtf(var + 1e-5f);
  }
  __syncthreads();
  for (int idx = threadIdx.x; idx < 32 * N; idx += blockDim.x) {
    int rr = idx / N, col = idx % N;
    float v = (tileC[idx] - stats[rr]) * stats[32 + rr] * lng[col] + lnb[col];
    xc[(size_t)(m0 + rr) * 1024 + colOff + col] = __float2bfloat16(gelu_exact(v));
  }
}

// x4 = GELU(LN(x)) -> xc[:, 512:1024]
__global__ void x4_kernel(const float* __restrict__ x, const float* __restrict__ g,
                          const float* __restrict__ bb, bf16* __restrict__ xc) {
  __shared__ float stats[64];
  const int m0 = blockIdx.x * 32;
  if (threadIdx.x < 32) {
    const float* row = x + (size_t)(m0 + threadIdx.x) * 512;
    float s = 0.f, ss = 0.f;
    for (int j = 0; j < 512; ++j) { float v = row[j]; s += v; ss += v * v; }
    float mean = s * (1.0f / 512.0f);
    float var  = ss * (1.0f / 512.0f) - mean * mean;
    stats[threadIdx.x]      = mean;
    stats[32 + threadIdx.x] = rsqrtf(var + 1e-5f);
  }
  __syncthreads();
  for (int idx = threadIdx.x; idx < 32 * 512; idx += blockDim.x) {
    int rr = idx >> 9, col = idx & 511;
    float v = x[(size_t)(m0 + rr) * 512 + col];
    v = (v - stats[rr]) * stats[32 + rr] * g[col] + bb[col];
    xc[(size_t)(m0 + rr) * 1024 + 512 + col] = __float2bfloat16(gelu_exact(v));
  }
}

// q = (x @ Wq^T) * SCALE -> bf16 [B,8,256,64]
__global__ void gemm_q(const bf16* __restrict__ A, const bf16* __restrict__ Wt,
                       bf16* __restrict__ qb) {
  const int m0   = blockIdx.x * 32;
  const int wave = threadIdx.x >> 5;
  const int lane = threadIdx.x & 31;
  const int hi   = lane >> 4;
  const int l15  = lane & 15;
  for (int t = wave; t < 16; t += 8) {          // 16 n-macros of 32 cols
    v8f acc[2][2] = {};
    gemm_2x2(A, 512, m0, Wt, 512, t * 32, 512, acc);
#pragma unroll
    for (int u = 0; u < 2; ++u)
#pragma unroll
      for (int vv = 0; vv < 2; ++vv)
#pragma unroll
        for (int i = 0; i < 8; ++i) {
          int m = m0 + u * 16 + i + 8 * hi;
          int n = t * 32 + vv * 16 + l15;
          int b = m >> 8, rr = m & 255, h = n >> 6, d = n & 63;
          qb[((size_t)(b * 8 + h) * 256 + rr) * 64 + d] =
              __float2bfloat16(acc[u][vv][i] * 0.125f);
        }
  }
}

// kv = xc @ Wkv^T ; n<512 -> k[B,8,256,64], n>=512 -> vT[B,8,64,256]
__global__ void gemm_kv(const bf16* __restrict__ A, const bf16* __restrict__ Wt,
                        bf16* __restrict__ kb, bf16* __restrict__ vT) {
  const int m0   = blockIdx.x * 32;
  const int wave = threadIdx.x >> 5;
  const int lane = threadIdx.x & 31;
  const int hi   = lane >> 4;
  const int l15  = lane & 15;
  for (int t = wave; t < 32; t += 8) {          // 32 n-macros of 32 cols
    v8f acc[2][2] = {};
    gemm_2x2(A, 1024, m0, Wt, 1024, t * 32, 1024, acc);
#pragma unroll
    for (int u = 0; u < 2; ++u)
#pragma unroll
      for (int vv = 0; vv < 2; ++vv)
#pragma unroll
        for (int i = 0; i < 8; ++i) {
          int m = m0 + u * 16 + i + 8 * hi;
          int n = t * 32 + vv * 16 + l15;
          int b = m >> 8, rr = m & 255;
          int kvsel = n >> 9, hn = n & 511, h = hn >> 6, d = hn & 63;
          bf16 w = __float2bfloat16(acc[u][vv][i]);
          if (kvsel == 0)
            kb[((size_t)(b * 8 + h) * 256 + rr) * 64 + d] = w;
          else
            vT[((size_t)(b * 8 + h) * 64 + d) * 256 + rr] = w;
        }
  }
}

// Attention: per (b,h,half): S = q k^T (scale pre-folded), softmax, O = P vT^T.
// S (f32, 128x256) + P (bf16, 128x256) live in 192 KB of dynamic LDS.
__global__ void attn_kernel(const bf16* __restrict__ qb, const bf16* __restrict__ kb,
                            const bf16* __restrict__ vT, bf16* __restrict__ ao) {
  extern __shared__ char smraw[];
  float* S = (float*)smraw;                       // 128*256*4 = 128 KB
  bf16*  P = (bf16*)(smraw + 128 * 256 * 4);      // 128*256*2 =  64 KB
  const int blk = blockIdx.x;
  const int half128 = blk & 1;
  const int h = (blk >> 1) & 7;
  const int b = blk >> 4;
  const bf16* q0 = qb + ((size_t)(b * 8 + h) * 256 + half128 * 128) * 64;
  const bf16* k0 = kb + (size_t)(b * 8 + h) * 256 * 64;
  const bf16* v0 = vT + (size_t)(b * 8 + h) * 64 * 256;
  const int wave = threadIdx.x >> 5;
  const int lane = threadIdx.x & 31;
  const int hi   = lane >> 4;
  const int l15  = lane & 15;
  // Phase 1: scores. 4 m-macros x 8 n-macros of 32x32.
  for (int t = wave; t < 32; t += 8) {
    int mt = t >> 3, nt = t & 7;
    v8f acc[2][2] = {};
    gemm_2x2(q0, 64, mt * 32, k0, 64, nt * 32, 64, acc);
#pragma unroll
    for (int u = 0; u < 2; ++u)
#pragma unroll
      for (int vv = 0; vv < 2; ++vv)
#pragma unroll
        for (int i = 0; i < 8; ++i)
          S[(mt * 32 + u * 16 + i + 8 * hi) * 256 + nt * 32 + vv * 16 + l15] =
              acc[u][vv][i];
  }
  __syncthreads();
  // Phase 2: row softmax -> bf16 P
  if (threadIdx.x < 128) {
    float* row = S + (size_t)threadIdx.x * 256;
    float mx = -3.4e38f;
    for (int j = 0; j < 256; ++j) mx = fmaxf(mx, row[j]);
    float sum = 0.f;
    for (int j = 0; j < 256; ++j) { float e = __expf(row[j] - mx); row[j] = e; sum += e; }
    float inv = 1.0f / sum;
    bf16* prow = P + (size_t)threadIdx.x * 256;
    for (int j = 0; j < 256; ++j) prow[j] = __float2bfloat16(row[j] * inv);
  }
  __syncthreads();
  // Phase 3: O = P @ vT^T. 4 m-macros x 2 d-macros of 32x32, K=256.
  for (int t = wave; t < 8; t += 8) {
    int mt = t >> 1, dt = t & 1;
    v8f acc[2][2] = {};
    gemm_2x2((const bf16*)P, 256, mt * 32, v0, 256, dt * 32, 256, acc);
#pragma unroll
    for (int u = 0; u < 2; ++u)
#pragma unroll
      for (int vv = 0; vv < 2; ++vv)
#pragma unroll
        for (int i = 0; i < 8; ++i) {
          int m = half128 * 128 + mt * 32 + u * 16 + i + 8 * hi;
          int d = dt * 32 + vv * 16 + l15;
          ao[((size_t)b * 256 + m) * 512 + h * 64 + d] =
              __float2bfloat16(acc[u][vv][i]);
        }
  }
}

// out = attn_out @ proj_w^T + proj_b  (f32 output)
__global__ void gemm_proj(const bf16* __restrict__ A, const bf16* __restrict__ Wt,
                          const float* __restrict__ bias, float* __restrict__ out) {
  const int m0   = blockIdx.x * 32;
  const int wave = threadIdx.x >> 5;
  const int lane = threadIdx.x & 31;
  const int hi   = lane >> 4;
  const int l15  = lane & 15;
  for (int t = wave; t < 16; t += 8) {
    v8f acc[2][2] = {};
    gemm_2x2(A, 512, m0, Wt, 512, t * 32, 512, acc);
#pragma unroll
    for (int u = 0; u < 2; ++u)
#pragma unroll
      for (int vv = 0; vv < 2; ++vv)
#pragma unroll
        for (int i = 0; i < 8; ++i) {
          int m = m0 + u * 16 + i + 8 * hi;
          int n = t * 32 + vv * 16 + l15;
          out[(size_t)m * 512 + n] = acc[u][vv][i] + bias[n];
        }
  }
}

// ---------------------------------------------------------------------------
extern "C" void kernel_launch(void* const* d_in, const int* in_sizes, int n_in,
                              void* d_out, int out_size, void* d_ws, size_t ws_size,
                              hipStream_t stream) {
  const float* x      = (const float*)d_in[0];
  const float* y1     = (const float*)d_in[1];
  const float* y2     = (const float*)d_in[2];
  const float* y3     = (const float*)d_in[3];
  const float* Wq     = (const float*)d_in[10];
  const float* Wkv    = (const float*)d_in[11];
  const float* sr1_w  = (const float*)d_in[12];
  const float* sr1_b  = (const float*)d_in[13];
  const float* ln1_g  = (const float*)d_in[14];
  const float* ln1_b  = (const float*)d_in[15];
  const float* sr2_w  = (const float*)d_in[16];
  const float* sr2_b  = (const float*)d_in[17];
  const float* ln2_g  = (const float*)d_in[18];
  const float* ln2_b  = (const float*)d_in[19];
  const float* sr3_w  = (const float*)d_in[20];
  const float* sr3_b  = (const float*)d_in[21];
  const float* ln3_g  = (const float*)d_in[22];
  const float* ln3_b  = (const float*)d_in[23];
  const float* ln4_g  = (const float*)d_in[24];
  const float* ln4_b  = (const float*)d_in[25];
  const float* proj_w = (const float*)d_in[26];
  const float* proj_b = (const float*)d_in[27];
  float* out = (float*)d_out;

  char* ws = (char*)d_ws;
  size_t off = 0;
  auto alloc = [&](size_t bytes) -> void* {
    void* p = ws + off;
    off += (bytes + 255) & ~(size_t)255;
    return p;
  };
  const size_t M = 32 * 256;   // 8192 token rows
  bf16* xbf     = (bf16*)alloc(M * 512 * 2);
  bf16* pooled1 = (bf16*)alloc(M * 64 * 2);
  bf16* pooled2 = (bf16*)alloc(M * 128 * 2);
  bf16* pooled3 = (bf16*)alloc(M * 320 * 2);
  bf16* xc      = (bf16*)alloc(M * 1024 * 2);
  bf16* qb      = (bf16*)alloc(M * 512 * 2);
  bf16* kb      = (bf16*)alloc(M * 512 * 2);
  bf16* vTb     = (bf16*)alloc(M * 512 * 2);
  bf16* aob     = (bf16*)alloc(M * 512 * 2);
  bf16* wq_bf   = (bf16*)alloc(512 * 512 * 2);
  bf16* wkv_bf  = (bf16*)alloc(1024 * 1024 * 2);
  bf16* sr1_bf  = (bf16*)alloc(64 * 64 * 2);
  bf16* sr2_bf  = (bf16*)alloc(128 * 128 * 2);
  bf16* sr3_bf  = (bf16*)alloc(320 * 320 * 2);
  bf16* wpj_bf  = (bf16*)alloc(512 * 512 * 2);

  auto cvt = [&](const float* s, bf16* d, int n) {
    convert_f32_bf16<<<(n + 255) / 256, 256, 0, stream>>>(s, d, n);
  };
  cvt(x, xbf, (int)(M * 512));
  cvt(Wq, wq_bf, 512 * 512);
  cvt(Wkv, wkv_bf, 1024 * 1024);
  cvt(sr1_w, sr1_bf, 64 * 64);
  cvt(sr2_w, sr2_bf, 128 * 128);
  cvt(sr3_w, sr3_bf, 320 * 320);
  cvt(proj_w, wpj_bf, 512 * 512);

  pool_kernel<<<8192, 256, 0, stream>>>(y1, pooled1, 128, 128, 64, 8);
  pool_kernel<<<8192, 256, 0, stream>>>(y2, pooled2, 64, 64, 128, 4);
  pool_kernel<<<8192, 256, 0, stream>>>(y3, pooled3, 32, 32, 320, 2);

  gemm_ln_gelu<<<256, 256, 32 * 64 * 4 + 256, stream>>>(pooled1, sr1_bf, sr1_b, ln1_g, ln1_b, xc, 0, 64);
  gemm_ln_gelu<<<256, 256, 32 * 128 * 4 + 256, stream>>>(pooled2, sr2_bf, sr2_b, ln2_g, ln2_b, xc, 64, 128);
  gemm_ln_gelu<<<256, 256, 32 * 320 * 4 + 256, stream>>>(pooled3, sr3_bf, sr3_b, ln3_g, ln3_b, xc, 192, 320);
  x4_kernel<<<256, 256, 0, stream>>>(x, ln4_g, ln4_b, xc);

  gemm_q<<<256, 256, 0, stream>>>(xbf, wq_bf, qb);
  gemm_kv<<<256, 256, 0, stream>>>(xc, wkv_bf, kb, vTb);

  attn_kernel<<<512, 256, 128 * 256 * 4 + 128 * 256 * 2, stream>>>(qb, kb, vTb, aob);

  gemm_proj<<<256, 256, 0, stream>>>(aob, wpj_bf, proj_b, out);
}
